// informativeness_18339510354595
// MI455X (gfx1250) — matrix operand order repaired and checked by reading, not compiled, not run on hardware
//
#include <hip/hip_runtime.h>
#include <math.h>

#define NMAT 8192
#define KDIM 768
#define KT   32     // WMMA K per step
#define BT   128    // block tile (M and N)
#define LDT  40     // padded LDS row stride in bf16 elems (80 B; 16B-aligned chunks, conflict-free)
#define PS   (BT * LDT)          // plane elems
#define PSB  (PS * 2)            // plane bytes (10240)
#define BUFB (4 * PSB)           // buffer bytes: Ahi,Alo,Bhi,Blo (40960)

typedef __attribute__((ext_vector_type(16))) __bf16 v16bf;
typedef __attribute__((ext_vector_type(8)))  float  v8f;

// ---- order-preserving float <-> uint encoding (for atomicMax on any-sign floats) ----
__device__ __forceinline__ unsigned encf(float f) {
    unsigned u = __float_as_uint(f);
    return (u & 0x80000000u) ? ~u : (u | 0x80000000u);
}
__device__ __forceinline__ float decf(unsigned e) {
    return (e & 0x80000000u) ? __uint_as_float(e ^ 0x80000000u)
                             : __uint_as_float(~e);
}

union FragU { v16bf v; uint4 q[2]; };
__device__ __forceinline__ v16bf make_frag(uint4 a, uint4 b) {
    FragU u; u.q[0] = a; u.q[1] = b; return u.v;
}

// ---- CDNA5 async global->LDS copy (ASYNCcnt-tracked; bypasses VGPRs) ----
__device__ __forceinline__ void async_b128(unsigned lds_byte_off, const void* gaddr) {
    asm volatile("global_load_async_to_lds_b128 %0, %1, off"
                 :: "v"(lds_byte_off), "v"((unsigned long long)gaddr)
                 : "memory");
}
__device__ __forceinline__ void wait_async0() {
    asm volatile("s_wait_asynccnt 0" ::: "memory");
}

// ---------------- stage 0: zero the encoded max buffers ----------------
__global__ __launch_bounds__(256) void init_max_kernel(unsigned* __restrict__ p, int n) {
    int i = blockIdx.x * 256 + threadIdx.x;
    if (i < n) p[i] = 0u;   // encoding 0 == smallest possible value
}

// ---------------- stage 1: row-normalize and split f32 -> bf16 hi/lo ----------------
__global__ __launch_bounds__(256) void normalize_rows_kernel(const float* __restrict__ src,
                                                             __bf16* __restrict__ hi,
                                                             __bf16* __restrict__ lo) {
    __shared__ float red[256];
    const int tid = threadIdx.x;
    const size_t base = (size_t)blockIdx.x * KDIM;
    float x0 = src[base + tid];
    float x1 = src[base + tid + 256];
    float x2 = src[base + tid + 512];
    red[tid] = x0 * x0 + x1 * x1 + x2 * x2;
    __syncthreads();
    for (int s = 128; s > 0; s >>= 1) {
        if (tid < s) red[tid] += red[tid + s];
        __syncthreads();
    }
    const float rinv = 1.0f / fmaxf(sqrtf(red[0]), 1e-8f);
    float v; __bf16 h;
    v = x0 * rinv; h = (__bf16)v; hi[base + tid      ] = h; lo[base + tid      ] = (__bf16)(v - (float)h);
    v = x1 * rinv; h = (__bf16)v; hi[base + tid + 256] = h; lo[base + tid + 256] = (__bf16)(v - (float)h);
    v = x2 * rinv; h = (__bf16)v; hi[base + tid + 512] = h; lo[base + tid + 512] = (__bf16)(v - (float)h);
}

// ---------------- stage 2: WMMA GEMM (C = Xn @ Yn^T) fused with row/col max ----------------
// Double-buffered LDS; next K-tile staged via async-to-LDS while WMMAs run on current tile.
__global__ __launch_bounds__(256) void gemm_max_kernel(const __bf16* __restrict__ Xhi,
                                                       const __bf16* __restrict__ Xlo,
                                                       const __bf16* __restrict__ Yhi,
                                                       const __bf16* __restrict__ Ylo,
                                                       unsigned* __restrict__ rowmax,
                                                       unsigned* __restrict__ colmax) {
    __shared__ __bf16 smem[2][4][PS];   // [buf][plane: Ahi,Alo,Bhi,Blo][row*LDT + k]
    __shared__ unsigned srow[BT];
    __shared__ unsigned scol[BT];

    const int tid  = threadIdx.x;
    const int lane = tid & 31;
    const int wave = tid >> 5;
    const int wm   = wave & 3;    // 4 waves down   -> 32 rows each
    const int wn   = wave >> 2;   // 2 waves across -> 64 cols each
    const int mrow = lane & 15;
    const int tileM = blockIdx.x * BT;
    const int tileN = blockIdx.y * BT;

    for (int i = tid; i < BT; i += 256) { srow[i] = 0u; scol[i] = 0u; }

    v8f acc[2][4];
#pragma unroll
    for (int mt = 0; mt < 2; ++mt)
#pragma unroll
        for (int nt = 0; nt < 4; ++nt)
            acc[mt][nt] = (v8f){0.f, 0.f, 0.f, 0.f, 0.f, 0.f, 0.f, 0.f};

    // ---- per-thread staging geometry: 512 16B chunks / plane, 2 per thread ----
    const int srw = tid >> 2;               // 0..63  (chunk rows; +64 for second chunk)
    const int kc  = (tid & 3) * 8;          // 0,8,16,24
    const __bf16* gxh = Xhi + (size_t)(tileM + srw) * KDIM + kc;
    const __bf16* gxl = Xlo + (size_t)(tileM + srw) * KDIM + kc;
    const __bf16* gyh = Yhi + (size_t)(tileN + srw) * KDIM + kc;
    const __bf16* gyl = Ylo + (size_t)(tileN + srw) * KDIM + kc;
    const size_t r64 = (size_t)64 * KDIM;   // +64 rows in global
    const unsigned ldsBase  = (unsigned)(size_t)&smem[0][0][0];          // LDS byte offset (addr[31:0])
    const unsigned ldsChunk = (unsigned)((srw * LDT + kc) * 2);          // within-plane byte offset
    const unsigned ldsRow64 = (unsigned)(64 * LDT * 2);

    // fragment addressing (ISA 16-bit A 16x32 / B 32x16 wave32 layouts)
    const int arsel = (lane < 16) ? 0 : 8;   // A: K 0-7&16-23 vs 8-15&24-31
    const int bksel = (lane < 16) ? 0 : 8;   // B: K 0-7&16-23 vs 8-15&24-31 of col N=lane%16

    const int NK = KDIM / KT;   // 24

    // ---- prologue: stage K-tile 0 into buffer 0 ----
    {
        const unsigned b = ldsBase + ldsChunk;
        async_b128(b + 0 * PSB,            gxh);
        async_b128(b + 0 * PSB + ldsRow64, gxh + r64);
        async_b128(b + 1 * PSB,            gxl);
        async_b128(b + 1 * PSB + ldsRow64, gxl + r64);
        async_b128(b + 2 * PSB,            gyh);
        async_b128(b + 2 * PSB + ldsRow64, gyh + r64);
        async_b128(b + 3 * PSB,            gyl);
        async_b128(b + 3 * PSB + ldsRow64, gyl + r64);
    }

    for (int kk = 0; kk < NK; ++kk) {
        const int cur = kk & 1;
        wait_async0();        // our async stages for buffer `cur` complete
        __syncthreads();      // all waves: stages visible, prior reads of other buffer done

        // ---- prefetch next K-tile into the other buffer (overlaps WMMA below) ----
        if (kk + 1 < NK) {
            const int k0n = (kk + 1) * KT;
            const unsigned b = ldsBase + (unsigned)(1 - cur) * BUFB + ldsChunk;
            async_b128(b + 0 * PSB,            gxh + k0n);
            async_b128(b + 0 * PSB + ldsRow64, gxh + k0n + r64);
            async_b128(b + 1 * PSB,            gxl + k0n);
            async_b128(b + 1 * PSB + ldsRow64, gxl + k0n + r64);
            async_b128(b + 2 * PSB,            gyh + k0n);
            async_b128(b + 2 * PSB + ldsRow64, gyh + k0n + r64);
            async_b128(b + 3 * PSB,            gyl + k0n);
            async_b128(b + 3 * PSB + ldsRow64, gyl + k0n + r64);
        }

        // ---- fragment loads from LDS ----
        const __bf16* sAh = &smem[cur][0][0];
        const __bf16* sAl = &smem[cur][1][0];
        const __bf16* sBh = &smem[cur][2][0];
        const __bf16* sBl = &smem[cur][3][0];

        v16bf afr[2][2], bfr[4][2];
#pragma unroll
        for (int mt = 0; mt < 2; ++mt) {
            const int r = (wm * 32 + mt * 16 + mrow) * LDT;
            afr[mt][0] = make_frag(*(const uint4*)&sAh[r + arsel],
                                   *(const uint4*)&sAh[r + 16 + arsel]);
            afr[mt][1] = make_frag(*(const uint4*)&sAl[r + arsel],
                                   *(const uint4*)&sAl[r + 16 + arsel]);
        }
#pragma unroll
        for (int nt = 0; nt < 4; ++nt) {
            const int r = (wn * 64 + nt * 16 + mrow) * LDT;
            bfr[nt][0] = make_frag(*(const uint4*)&sBh[r + bksel],
                                   *(const uint4*)&sBh[r + bksel + 16]);
            bfr[nt][1] = make_frag(*(const uint4*)&sBl[r + bksel],
                                   *(const uint4*)&sBl[r + bksel + 16]);
        }

        // ---- hi*hi + hi*lo + lo*hi : 24 WMMAs per wave per K-step ----
#pragma unroll
        for (int mt = 0; mt < 2; ++mt)
#pragma unroll
            for (int nt = 0; nt < 4; ++nt) {
                acc[mt][nt] = __builtin_amdgcn_wmma_f32_16x16x32_bf16(
                    false, afr[mt][0], false, bfr[nt][0], (short)0, acc[mt][nt], false, false);
                acc[mt][nt] = __builtin_amdgcn_wmma_f32_16x16x32_bf16(
                    false, afr[mt][0], false, bfr[nt][1], (short)0, acc[mt][nt], false, false);
                acc[mt][nt] = __builtin_amdgcn_wmma_f32_16x16x32_bf16(
                    false, afr[mt][1], false, bfr[nt][0], (short)0, acc[mt][nt], false, false);
            }
    }

    // ---- fused epilogue: per-lane partial maxima -> LDS atomics -> global atomics ----
    // C/D layout: VGPR i, lane L -> row = i (L<16) or 8+i (L>=16), col = L%16
    const int rhalf = (lane < 16) ? 0 : 8;
#pragma unroll
    for (int mt = 0; mt < 2; ++mt) {
        float rmx[8];
#pragma unroll
        for (int i = 0; i < 8; ++i) rmx[i] = -3.4e38f;
#pragma unroll
        for (int nt = 0; nt < 4; ++nt)
#pragma unroll
            for (int i = 0; i < 8; ++i) rmx[i] = fmaxf(rmx[i], acc[mt][nt][i]);
        const int rb = wm * 32 + mt * 16 + rhalf;
#pragma unroll
        for (int i = 0; i < 8; ++i) atomicMax(&srow[rb + i], encf(rmx[i]));
    }
#pragma unroll
    for (int nt = 0; nt < 4; ++nt) {
        float cmx = -3.4e38f;
#pragma unroll
        for (int mt = 0; mt < 2; ++mt)
#pragma unroll
            for (int i = 0; i < 8; ++i) cmx = fmaxf(cmx, acc[mt][nt][i]);
        atomicMax(&scol[wn * 64 + nt * 16 + mrow], encf(cmx));
    }
    __syncthreads();
    for (int i = tid; i < BT; i += 256) {
        atomicMax(&rowmax[tileM + i], srow[i]);
        atomicMax(&colmax[tileN + i], scol[i]);
    }
}

// ---------------- stage 3: -sum(logprob(1 + max)) for rows (out[0]) and cols (out[1]) ----------------
__global__ __launch_bounds__(256) void finalize_kernel(const unsigned* __restrict__ rowmax,
                                                       const unsigned* __restrict__ colmax,
                                                       float* __restrict__ out) {
    __shared__ float red[256];
    const unsigned* src = (blockIdx.x == 0) ? rowmax : colmax;
    const float LOGZ = logf(0.3f * 2.50662827463100050242f);  // log(sigma*sqrt(2*pi))
    float s = 0.0f;
    for (int i = threadIdx.x; i < NMAT; i += 256) {
        float c = decf(src[i]);                 // (1+c) - 1 = c
        float t = c * (1.0f / 0.3f);
        s += 0.5f * t * t + LOGZ;               // -log N(1+c; 1, sigma)
    }
    red[threadIdx.x] = s;
    __syncthreads();
    for (int st = 128; st > 0; st >>= 1) {
        if (threadIdx.x < st) red[threadIdx.x] += red[threadIdx.x + st];
        __syncthreads();
    }
    if (threadIdx.x == 0) out[blockIdx.x] = red[0];
}

extern "C" void kernel_launch(void* const* d_in, const int* in_sizes, int n_in,
                              void* d_out, int out_size, void* d_ws, size_t ws_size,
                              hipStream_t stream) {
    const float* ex = (const float*)d_in[0];
    const float* ey = (const float*)d_in[1];
    float* out = (float*)d_out;

    const size_t plane = (size_t)NMAT * KDIM;    // elements per bf16 plane
    __bf16* Xhi = (__bf16*)d_ws;
    __bf16* Xlo = Xhi + plane;
    __bf16* Yhi = Xlo + plane;
    __bf16* Ylo = Yhi + plane;
    unsigned* rowmax = (unsigned*)(Ylo + plane);
    unsigned* colmax = rowmax + NMAT;

    init_max_kernel<<<dim3((2 * NMAT + 255) / 256), dim3(256), 0, stream>>>(rowmax, 2 * NMAT);
    normalize_rows_kernel<<<dim3(NMAT), dim3(256), 0, stream>>>(ex, Xhi, Xlo);
    normalize_rows_kernel<<<dim3(NMAT), dim3(256), 0, stream>>>(ey, Yhi, Ylo);
    gemm_max_kernel<<<dim3(NMAT / BT, NMAT / BT), dim3(256), 0, stream>>>(
        Xhi, Xlo, Yhi, Ylo, rowmax, colmax);
    finalize_kernel<<<dim3(2), dim3(256), 0, stream>>>(rowmax, colmax, out);
}
